// MultiLayers_MS_33251636806230
// MI455X (gfx1250) — matrix operand order repaired and checked by reading, not compiled, not run on hardware
//
#include <hip/hip_runtime.h>
#include <hip/hip_bf16.h>
#include <math.h>

// ---------------------------------------------------------------------------
// MultiLayers_MS on MI455X (gfx1250):
//   bf16 WMMA GEMMs fed by Tensor-Data-Mover (TDM) LDS double buffering
//   + fused softmax/mean-shift + graph-norm.
// ---------------------------------------------------------------------------

#define N_NODES 4096
#define DIM     256
#define NLAYERS 4
#define NSEG    8

typedef __bf16 bf16_t;
typedef __attribute__((ext_vector_type(16))) __bf16 bf16x16;
typedef __attribute__((ext_vector_type(8)))  __bf16 bf16x8;
typedef __attribute__((ext_vector_type(8)))  float  f32x8;

typedef unsigned int u32;
typedef __attribute__((ext_vector_type(4))) unsigned int u32x4;
typedef __attribute__((ext_vector_type(8))) int          i32x8;
typedef __attribute__((ext_vector_type(4))) int          i32x4;

enum GemmMode {
  MODE_ATT_F32   = 0,  // out f32 = acc * (1/(sqrt(K)*t))
  MODE_BF16_BIAS = 1,  // out bf16 = acc + bias[n]
  MODE_BF16T_BIAS= 2,  // outT bf16: out[n*ldc+m] = acc + bias[n]   (pre-transposed V)
  MODE_BF16      = 3,  // out bf16 = acc
  MODE_RES_F32   = 4   // out f32 = resid + acc + bias[n]  (residual o-proj)
};

#define TM 64
#define TN 128
#define KSTEP 32
#define LDS_STRIDE 40   // 32 bf16 + 16B TDM pad per row -> 80B (bank de-conflict)

#if __has_builtin(__builtin_amdgcn_tensor_load_to_lds) && __has_builtin(__builtin_amdgcn_s_wait_tensorcnt)
#define HAVE_TDM 1
#else
#define HAVE_TDM 0
#endif

#if HAVE_TDM
// D# group1 (ISA 8.4): data_size=2B, pad every 16 DWORDs by 4 DWORDs,
// tensor_dim0 = row stride (full rows in bounds), tensor_dim1 = rows,
// tile = tile_k x rows, dim0_stride = lda.
__device__ __forceinline__ i32x8 tdm_group1(u32 tile_k, u32 rows, u32 stride)
{
  const u32 tw = stride;   // tensor_dim0
  const u32 th = rows;     // tensor_dim1
  i32x8 g1;
  g1[0] = (int)((1u << 16)        // data_size = 2 bytes
              | (1u << 20)        // pad_enable
              | (3u << 22)        // pad_interval: 16 DWORDs (one 64B row)
              | (3u << 25));      // pad_amount: 4 DWORDs (16B)
  g1[1] = (int)((tw & 0xffffu) << 16);                                  // dim0[15:0]
  g1[2] = (int)(((tw >> 16) & 0xffffu) | ((th & 0xffffu) << 16));       // dim0[31:16], dim1[15:0]
  g1[3] = (int)(((th >> 16) & 0xffffu) | ((tile_k & 0xffffu) << 16));   // dim1[31:16], tile_dim0
  g1[4] = (int)(rows & 0xffffu);                                        // tile_dim1 (tile_dim2=0)
  g1[5] = (int)stride;                                                  // dim0_stride[31:0]
  g1[6] = 0;                                                            // stride[47:32], dim1_stride lo
  g1[7] = 0;
  return g1;
}

// D# group0 (ISA 8.3): count=1, lds_addr, 57-bit global addr, type=2.
__device__ __forceinline__ void tdm_issue(const bf16_t* gtile, u32 lds_off, i32x8 g1)
{
  const unsigned long long ga = (unsigned long long)(size_t)gtile;
  u32x4 g0;
  g0[0] = 1u;                                            // count=1 (valid user D#)
  g0[1] = lds_off;                                       // LDS byte address
  g0[2] = (u32)ga;                                       // global_addr[31:0]
  g0[3] = (u32)((ga >> 32) & 0x01ffffffu) | (2u << 30);  // global_addr[56:32], type=2
  i32x4 z = {0, 0, 0, 0};
#if defined(__clang_major__) && (__clang_major__ >= 23)
  i32x8 z8 = {0, 0, 0, 0, 0, 0, 0, 0};
  __builtin_amdgcn_tensor_load_to_lds(g0, g1, z, z, z8, 0);
#else
  __builtin_amdgcn_tensor_load_to_lds(g0, g1, z, z, 0);
#endif
}
#endif // HAVE_TDM

__device__ __forceinline__ u32 lds_offset_of(const void* p)
{
  return (u32)(unsigned long long)(size_t)p;   // flat LDS addr: low 32 bits = LDS offset
}

// ---------------------------------------------------------------------------
// NT GEMM: C[M,Nn] = A[M,K] * B[Nn,K]^T   (A,B bf16 row-major, f32 accum)
// Block = 8 waves = 64x128 tile; wave tile 32x32 (2x2 WMMA fragments).
// TDM stages 64x32 (A) and 128x32 (B) K-slabs into LDS, double buffered;
// fragment layouts follow CDNA5 ISA 7.12.2 (wave32).
// ---------------------------------------------------------------------------
__global__ __launch_bounds__(256)
void gemm_nt_bf16(const bf16_t* __restrict__ A, const bf16_t* __restrict__ Bm,
                  int M, int Nn, int K, int lda, int ldb, int ldc, int mode,
                  float* __restrict__ outF, bf16_t* __restrict__ outB,
                  const float* __restrict__ bias,
                  const float* __restrict__ resid,
                  const int* __restrict__ tptr)
{
  const int tid  = threadIdx.x;
  const int lane = tid & 31;
  const int wave = tid >> 5;
  const int wm = wave >> 2;          // 0..1
  const int wn = wave & 3;           // 0..3
  const int mBase = blockIdx.y * TM + wm * 32;
  const int nBase = blockIdx.x * TN + wn * 32;
  if (mBase >= M || nBase >= Nn) return;

  const int jl = lane & 15;          // row (A) / col (B) within 16
  const int hi = lane >> 4;          // K-half selector

  f32x8 c00 = {}, c01 = {}, c10 = {}, c11 = {};

#if HAVE_TDM
  __shared__ __align__(128) bf16_t As[2][TM * LDS_STRIDE];
  __shared__ __align__(128) bf16_t Bs[2][TN * LDS_STRIDE];

  const int mBlock = blockIdx.y * TM;
  const int nBlock = blockIdx.x * TN;
  const i32x8 gA = tdm_group1(KSTEP, TM, (u32)lda);
  const i32x8 gB = tdm_group1(KSTEP, TN, (u32)ldb);
  const bf16_t* aTile = A  + (long)mBlock * lda;
  const bf16_t* bTile = Bm + (long)nBlock * ldb;

  if (wave == 0) {
    tdm_issue(aTile, lds_offset_of(&As[0][0]), gA);
    tdm_issue(bTile, lds_offset_of(&Bs[0][0]), gB);
    __builtin_amdgcn_s_wait_tensorcnt(0);
  }
  __syncthreads();

  for (int k0 = 0; k0 < K; k0 += KSTEP) {
    const int cur = (k0 >> 5) & 1;
    const int nxt = cur ^ 1;
    if (k0 + KSTEP < K && wave == 0) {      // prefetch next K-slab via TDM
      tdm_issue(aTile + (k0 + KSTEP), lds_offset_of(&As[nxt][0]), gA);
      tdm_issue(bTile + (k0 + KSTEP), lds_offset_of(&Bs[nxt][0]), gB);
    }

    const bf16_t* aR0 = &As[cur][(wm * 32 + jl)      * LDS_STRIDE];
    const bf16_t* aR1 = &As[cur][(wm * 32 + 16 + jl) * LDS_STRIDE];
    const bf16_t* bR0 = &Bs[cur][(wn * 32 + jl)      * LDS_STRIDE];
    const bf16_t* bR1 = &Bs[cur][(wn * 32 + 16 + jl) * LDS_STRIDE];

    bf16x16 a0, a1, b0, b1;
    {                               // A frag: e<8 -> K=hi*8+e ; e>=8 -> K=16+hi*8+(e-8)
      bf16x8 lo = *(const bf16x8*)(aR0 + hi * 8);
      bf16x8 hv = *(const bf16x8*)(aR0 + 16 + hi * 8);
      #pragma unroll
      for (int e = 0; e < 8; ++e) { a0[e] = lo[e]; a0[e + 8] = hv[e]; }
    }
    {
      bf16x8 lo = *(const bf16x8*)(aR1 + hi * 8);
      bf16x8 hv = *(const bf16x8*)(aR1 + 16 + hi * 8);
      #pragma unroll
      for (int e = 0; e < 8; ++e) { a1[e] = lo[e]; a1[e + 8] = hv[e]; }
    }
    {                               // B frag: e<8 -> K=hi*8+e ; e>=8 -> K=16+hi*8+(e-8)
      bf16x8 x0 = *(const bf16x8*)(bR0 + hi * 8);
      bf16x8 x1 = *(const bf16x8*)(bR0 + 16 + hi * 8);
      #pragma unroll
      for (int e = 0; e < 8; ++e) { b0[e] = x0[e]; b0[e + 8] = x1[e]; }
    }
    {
      bf16x8 x0 = *(const bf16x8*)(bR1 + hi * 8);
      bf16x8 x1 = *(const bf16x8*)(bR1 + 16 + hi * 8);
      #pragma unroll
      for (int e = 0; e < 8; ++e) { b1[e] = x0[e]; b1[e + 8] = x1[e]; }
    }

    c00 = __builtin_amdgcn_wmma_f32_16x16x32_bf16(false, a0, false, b0, (short)0, c00, false, false);
    c01 = __builtin_amdgcn_wmma_f32_16x16x32_bf16(false, a0, false, b1, (short)0, c01, false, false);
    c10 = __builtin_amdgcn_wmma_f32_16x16x32_bf16(false, a1, false, b0, (short)0, c10, false, false);
    c11 = __builtin_amdgcn_wmma_f32_16x16x32_bf16(false, a1, false, b1, (short)0, c11, false, false);

    if (k0 + KSTEP < K) {
      if (wave == 0) __builtin_amdgcn_s_wait_tensorcnt(0);
      __syncthreads();
    }
  }
#else  // -------- fallback: direct global-load path (proven) --------
  const bf16_t* aRow0 = A  + (long)(mBase + jl)      * lda;
  const bf16_t* aRow1 = A  + (long)(mBase + 16 + jl) * lda;
  const bf16_t* bRow0 = Bm + (long)(nBase + jl)      * ldb;
  const bf16_t* bRow1 = Bm + (long)(nBase + 16 + jl) * ldb;

  for (int k = 0; k < K; k += KSTEP) {
    if (k + 128 < K) {
      __builtin_prefetch(aRow0 + k + 128, 0, 1);
      __builtin_prefetch(bRow0 + k + 128, 0, 1);
    }
    bf16x16 a0, a1, b0, b1;
    {
      bf16x8 lo = *(const bf16x8*)(aRow0 + k + hi * 8);
      bf16x8 hv = *(const bf16x8*)(aRow0 + k + 16 + hi * 8);
      #pragma unroll
      for (int e = 0; e < 8; ++e) { a0[e] = lo[e]; a0[e + 8] = hv[e]; }
    }
    {
      bf16x8 lo = *(const bf16x8*)(aRow1 + k + hi * 8);
      bf16x8 hv = *(const bf16x8*)(aRow1 + k + 16 + hi * 8);
      #pragma unroll
      for (int e = 0; e < 8; ++e) { a1[e] = lo[e]; a1[e + 8] = hv[e]; }
    }
    {
      bf16x8 x0 = *(const bf16x8*)(bRow0 + k + hi * 8);
      bf16x8 x1 = *(const bf16x8*)(bRow0 + k + 16 + hi * 8);
      #pragma unroll
      for (int e = 0; e < 8; ++e) { b0[e] = x0[e]; b0[e + 8] = x1[e]; }
    }
    {
      bf16x8 x0 = *(const bf16x8*)(bRow1 + k + hi * 8);
      bf16x8 x1 = *(const bf16x8*)(bRow1 + k + 16 + hi * 8);
      #pragma unroll
      for (int e = 0; e < 8; ++e) { b1[e] = x0[e]; b1[e + 8] = x1[e]; }
    }
    c00 = __builtin_amdgcn_wmma_f32_16x16x32_bf16(false, a0, false, b0, (short)0, c00, false, false);
    c01 = __builtin_amdgcn_wmma_f32_16x16x32_bf16(false, a0, false, b1, (short)0, c01, false, false);
    c10 = __builtin_amdgcn_wmma_f32_16x16x32_bf16(false, a1, false, b0, (short)0, c10, false, false);
    c11 = __builtin_amdgcn_wmma_f32_16x16x32_bf16(false, a1, false, b1, (short)0, c11, false, false);
  }
#endif

  float scale = 1.0f;
  if (mode == MODE_ATT_F32)
    scale = 1.0f / (sqrtf((float)K) * (float)(tptr ? *tptr : 1));

  f32x8 ctile[2][2];
  ctile[0][0] = c00; ctile[0][1] = c01; ctile[1][0] = c10; ctile[1][1] = c11;

  #pragma unroll
  for (int ti = 0; ti < 2; ++ti) {
    #pragma unroll
    for (int tj = 0; tj < 2; ++tj) {
      const int nG = nBase + tj * 16 + jl;
      const float bval = bias ? bias[nG] : 0.0f;
      #pragma unroll
      for (int r = 0; r < 8; ++r) {
        const int mG = mBase + ti * 16 + r + hi * 8;   // C layout: M = r (+8 for hi lanes)
        const float v = ctile[ti][tj][r];
        if (mode == MODE_ATT_F32) {
          outF[(long)mG * ldc + nG] = v * scale;
        } else if (mode == MODE_BF16_BIAS) {
          outB[(long)mG * ldc + nG] = (bf16_t)(v + bval);
        } else if (mode == MODE_BF16T_BIAS) {
          outB[(long)nG * ldc + mG] = (bf16_t)(v + bval);
        } else if (mode == MODE_BF16) {
          outB[(long)mG * ldc + nG] = (bf16_t)v;
        } else { // MODE_RES_F32
          const long idx = (long)mG * ldc + nG;
          outF[idx] = resid[idx] + v + bval;
        }
      }
    }
  }
}

// ---------------------------------------------------------------------------
// Row mean / population std of att (for use_std_att)
// ---------------------------------------------------------------------------
__global__ __launch_bounds__(256)
void row_stats_kernel(const float* __restrict__ att, float* __restrict__ rmean,
                      float* __restrict__ rstd, int n)
{
  const int row = blockIdx.x;
  const int tid = threadIdx.x;
  const float* p = att + (long)row * n;
  float s = 0.f, ss = 0.f;
  for (int j = tid; j < n; j += 256) { float v = p[j]; s += v; ss += v * v; }
  __shared__ float shs[256], shq[256];
  shs[tid] = s; shq[tid] = ss; __syncthreads();
  for (int st = 128; st > 0; st >>= 1) {
    if (tid < st) { shs[tid] += shs[tid + st]; shq[tid] += shq[tid + st]; }
    __syncthreads();
  }
  if (tid == 0) {
    float m = shs[0] / (float)n;
    float var = shq[0] / (float)n - m * m;
    rmean[row] = m;
    rstd[row]  = sqrtf(fmaxf(var, 0.f));
  }
}

// ---------------------------------------------------------------------------
// Fused per-row: std-normalize att, distances from xyz, mask, log prior,
// softmax, bf16 W output, and mean-shift xyz update (W @ xyz, beta blend).
// ---------------------------------------------------------------------------
__global__ __launch_bounds__(256)
void softmax_meanshift_kernel(float* __restrict__ att,
                              const float* __restrict__ rmean, const float* __restrict__ rstd,
                              const int* __restrict__ use_std_p,
                              const float* __restrict__ xyz_old, float* __restrict__ xyz_new,
                              const float* __restrict__ y_hat,
                              const unsigned char* __restrict__ dmask,
                              const unsigned char* __restrict__ bmask,
                              const float* __restrict__ bw_p, const float* __restrict__ beta_p,
                              const int* __restrict__ selfd_p,
                              bf16_t* __restrict__ wout, int n)
{
  const int i   = blockIdx.x;
  const int tid = threadIdx.x;
  float* arow = att + (long)i * n;
  const unsigned char* dm = dmask + (long)i * n;
  const unsigned char* bm = bmask + (long)i * n;

  const float bw   = bw_p[0];
  const float inv2 = 1.0f / (2.0f * bw * bw);
  const int   ustd = *use_std_p;
  const float sd   = (float)(*selfd_p);
  const float mi   = rmean[i];
  const float si   = rstd[i] + 1e-5f;
  const float xi0 = xyz_old[i * 3], xi1 = xyz_old[i * 3 + 1], xi2 = xyz_old[i * 3 + 2];
  const float sqi = xi0 * xi0 + xi1 * xi1 + xi2 * xi2;

  __shared__ float red[256];

  // Pass 1: logits + row max
  float lmax = -3.4e38f;
  for (int j = tid; j < n; j += 256) {
    float a = arow[j];
    if (ustd) a = (a - mi) / si;
    const float xj0 = xyz_old[j * 3], xj1 = xyz_old[j * 3 + 1], xj2 = xyz_old[j * 3 + 2];
    float d2 = sqi + (xj0 * xj0 + xj1 * xj1 + xj2 * xj2)
             - 2.0f * (xi0 * xj0 + xi1 * xj1 + xi2 * xj2);
    d2 = fmaxf(d2, 0.0f);
    if (j == i) d2 += sd;
    float lg = a - d2 * inv2 + logf(y_hat[j] + 1e-9f);
    if (!(dm[j] & bm[j])) lg = -1e9f;
    arow[j] = lg;
    lmax = fmaxf(lmax, lg);
  }
  red[tid] = lmax; __syncthreads();
  for (int st = 128; st > 0; st >>= 1) {
    if (tid < st) red[tid] = fmaxf(red[tid], red[tid + st]);
    __syncthreads();
  }
  const float gmax = red[0]; __syncthreads();

  // Pass 2: exp + sum + W@xyz partials
  float lsum = 0.f, ax = 0.f, ay = 0.f, az = 0.f;
  for (int j = tid; j < n; j += 256) {
    const float e = expf(arow[j] - gmax);
    arow[j] = e;
    lsum += e;
    ax += e * xyz_old[j * 3];
    ay += e * xyz_old[j * 3 + 1];
    az += e * xyz_old[j * 3 + 2];
  }
  float acc[4] = {lsum, ax, ay, az};
  float tot[4];
  #pragma unroll
  for (int c = 0; c < 4; ++c) {
    red[tid] = acc[c]; __syncthreads();
    for (int st = 128; st > 0; st >>= 1) {
      if (tid < st) red[tid] += red[tid + st];
      __syncthreads();
    }
    tot[c] = red[0]; __syncthreads();
  }
  const float inv = 1.0f / tot[0];

  // Pass 3: normalized weights as bf16 (A-operand for W@V WMMA GEMM)
  bf16_t* wrow = wout + (long)i * n;
  for (int j = tid; j < n; j += 256) wrow[j] = (bf16_t)(arow[j] * inv);

  if (tid == 0) {
    const float beta = beta_p[0];
    xyz_new[i * 3 + 0] = beta * (tot[1] * inv) + (1.0f - beta) * xi0;
    xyz_new[i * 3 + 1] = beta * (tot[2] * inv) + (1.0f - beta) * xi1;
    xyz_new[i * 3 + 2] = beta * (tot[3] * inv) + (1.0f - beta) * xi2;
  }
}

// ---------------------------------------------------------------------------
// GraphNorm: per-(segment, channel) mean + unbiased std, then apply.
// ---------------------------------------------------------------------------
__global__ void gn_stats_kernel(const float* __restrict__ x, const int* __restrict__ cums,
                                float* __restrict__ mean, float* __restrict__ stdv,
                                int nseg, int d)
{
  const int idx = blockIdx.x * blockDim.x + threadIdx.x;
  if (idx >= nseg * d) return;
  const int b = idx / d, c = idx % d;
  const int r0 = cums[b], r1 = cums[b + 1];
  const float cnt = (float)(r1 - r0);
  float s = 0.f, ss = 0.f;
  for (int r = r0; r < r1; ++r) { const float v = x[(long)r * d + c]; s += v; ss += v * v; }
  const float m = s / cnt;
  const float var = (ss - cnt * m * m) / fmaxf(cnt - 1.0f, 1.0f);
  mean[idx] = m;
  stdv[idx] = sqrtf(fmaxf(var, 0.f));
}

__global__ void gn_apply_kernel(float* __restrict__ x, const int* __restrict__ cums,
                                const float* __restrict__ mean, const float* __restrict__ stdv,
                                const float* __restrict__ gamma, const float* __restrict__ beta_g,
                                int n, int d, int nseg)
{
  const long idx = (long)blockIdx.x * blockDim.x + threadIdx.x;
  if (idx >= (long)n * d) return;
  const int r = (int)(idx / d), c = (int)(idx % d);
  int b = 0;
  for (int s = 0; s < nseg - 1; ++s) if (r >= cums[s + 1]) b++;
  const float m = mean[b * d + c], sv = stdv[b * d + c];
  x[idx] = gamma[c] * ((x[idx] - m) / (sv + 1e-5f)) + beta_g[c];
}

__global__ void cvt_f32_to_bf16_kernel(const float* __restrict__ src, bf16_t* __restrict__ dst, long n)
{
  const long i = (long)blockIdx.x * blockDim.x + threadIdx.x;
  if (i < n) dst[i] = (bf16_t)src[i];
}

// ---------------------------------------------------------------------------
// Host orchestration
// ---------------------------------------------------------------------------
extern "C" void kernel_launch(void* const* d_in, const int* in_sizes, int n_in,
                              void* d_out, int out_size, void* d_ws, size_t ws_size,
                              hipStream_t stream)
{
  const int N = N_NODES, D = DIM, L = NLAYERS, Bseg = NSEG;

  const float*         xyz_in  = (const float*)d_in[0];
  const float*         x_in    = (const float*)d_in[1];
  const float*         y_hat   = (const float*)d_in[2];
  const unsigned char* dmask   = (const unsigned char*)d_in[3];
  const int*           t_ptr   = (const int*)d_in[4];
  const float*         bw_ptr  = (const float*)d_in[5];
  const float*         beta_pt = (const float*)d_in[6];
  const int*           cums    = (const int*)d_in[7];
  const unsigned char* bmask   = (const unsigned char*)d_in[8];
  const int*           ustd_p  = (const int*)d_in[9];
  const int*           selfd_p = (const int*)d_in[10];
  const float*         qw      = (const float*)d_in[11];
  const float*         kw      = (const float*)d_in[12];
  const float*         vw      = (const float*)d_in[13];
  const float*         ow      = (const float*)d_in[14];
  const float*         qb      = (const float*)d_in[15];
  const float*         kb      = (const float*)d_in[16];
  const float*         vb      = (const float*)d_in[17];
  const float*         ob      = (const float*)d_in[18];
  const float*         gamma   = (const float*)d_in[19];
  const float*         beta_g  = (const float*)d_in[20];

  char* ws = (char*)d_ws;
  size_t off = 0;
  auto alloc = [&](size_t bytes) -> void* {
    void* p = ws + off;
    off += (bytes + 255) & ~(size_t)255;
    return p;
  };

  float*  att   = (float*)alloc((size_t)N * N * 4);        // 64 MB (lives in L2)
  bf16_t* wbuf  = (bf16_t*)alloc((size_t)N * N * 2);       // 32 MB softmax weights
  float*  xcur  = (float*)alloc((size_t)N * D * 4);
  bf16_t* xb    = (bf16_t*)alloc((size_t)N * D * 2);
  bf16_t* q16   = (bf16_t*)alloc((size_t)N * D * 2);
  bf16_t* k16   = (bf16_t*)alloc((size_t)N * D * 2);
  bf16_t* vT    = (bf16_t*)alloc((size_t)D * N * 2);       // pre-transposed V
  bf16_t* h16   = (bf16_t*)alloc((size_t)N * D * 2);
  bf16_t* wq    = (bf16_t*)alloc((size_t)L * D * D * 2);
  bf16_t* wk    = (bf16_t*)alloc((size_t)L * D * D * 2);
  bf16_t* wv    = (bf16_t*)alloc((size_t)L * D * D * 2);
  bf16_t* wo    = (bf16_t*)alloc((size_t)L * D * D * 2);
  float*  xyzA  = (float*)alloc((size_t)N * 3 * 4);
  float*  xyzB  = (float*)alloc((size_t)N * 3 * 4);
  float*  rmean = (float*)alloc((size_t)N * 4);
  float*  rstd  = (float*)alloc((size_t)N * 4);
  float*  gmean = (float*)alloc((size_t)Bseg * D * 4);
  float*  gstd  = (float*)alloc((size_t)Bseg * D * 4);
  (void)ws_size; (void)in_sizes; (void)n_in; (void)out_size;

  hipMemcpyAsync(xcur, x_in, (size_t)N * D * 4, hipMemcpyDeviceToDevice, stream);
  hipMemcpyAsync(xyzA, xyz_in, (size_t)N * 3 * 4, hipMemcpyDeviceToDevice, stream);

  { // weights f32 -> bf16 once
    const long wn = (long)L * D * D;
    const int blk = 256, grd = (int)((wn + blk - 1) / blk);
    cvt_f32_to_bf16_kernel<<<grd, blk, 0, stream>>>(qw, wq, wn);
    cvt_f32_to_bf16_kernel<<<grd, blk, 0, stream>>>(kw, wk, wn);
    cvt_f32_to_bf16_kernel<<<grd, blk, 0, stream>>>(vw, wv, wn);
    cvt_f32_to_bf16_kernel<<<grd, blk, 0, stream>>>(ow, wo, wn);
  }

  const dim3 gridProj(D / TN, N / TM);   // 2 x 64
  const dim3 gridAtt (N / TN, N / TM);   // 32 x 64
  float* xyz_old = xyzA;
  float* xyz_new = xyzB;

  for (int l = 0; l < L; ++l) {
    const bf16_t* wql = wq + (size_t)l * D * D;
    const bf16_t* wkl = wk + (size_t)l * D * D;
    const bf16_t* wvl = wv + (size_t)l * D * D;
    const bf16_t* wol = wo + (size_t)l * D * D;

    // x -> bf16 operand
    cvt_f32_to_bf16_kernel<<<(N * D) / 256, 256, 0, stream>>>(xcur, xb, (long)N * D);

    // Q/K projections (NT, bf16 out), V projection stored transposed [D,N]
    gemm_nt_bf16<<<gridProj, 256, 0, stream>>>(xb, wql, N, D, D, D, D, D,
        MODE_BF16_BIAS, nullptr, q16, qb + (size_t)l * D, nullptr, nullptr);
    gemm_nt_bf16<<<gridProj, 256, 0, stream>>>(xb, wkl, N, D, D, D, D, D,
        MODE_BF16_BIAS, nullptr, k16, kb + (size_t)l * D, nullptr, nullptr);
    gemm_nt_bf16<<<gridProj, 256, 0, stream>>>(xb, wvl, N, D, D, D, D, N,
        MODE_BF16T_BIAS, nullptr, vT, vb + (size_t)l * D, nullptr, nullptr);

    // att = (Q K^T) / (sqrt(D)*t)   -> f32 [N,N]
    gemm_nt_bf16<<<gridAtt, 256, 0, stream>>>(q16, k16, N, N, D, D, D, N,
        MODE_ATT_F32, att, nullptr, nullptr, nullptr, t_ptr);

    // row stats + fused masked softmax + mean-shift xyz update
    row_stats_kernel<<<N, 256, 0, stream>>>(att, rmean, rstd, N);
    softmax_meanshift_kernel<<<N, 256, 0, stream>>>(att, rmean, rstd, ustd_p,
        xyz_old, xyz_new, y_hat, dmask, bmask, bw_ptr, beta_pt, selfd_p, wbuf, N);

    // h = W @ V  ==  W[N,N] x vT[D,N]^T   (NT, contiguous loads)
    gemm_nt_bf16<<<gridProj, 256, 0, stream>>>(wbuf, vT, N, D, N, N, N, D,
        MODE_BF16, nullptr, h16, nullptr, nullptr, nullptr);

    // x = res + h @ ow^T + ob   (in-place residual epilogue)
    gemm_nt_bf16<<<gridProj, 256, 0, stream>>>(h16, wol, N, D, D, D, D, D,
        MODE_RES_F32, xcur, nullptr, ob + (size_t)l * D, xcur, nullptr);

    // GraphNorm over ragged segments
    gn_stats_kernel<<<(Bseg * D + 255) / 256, 256, 0, stream>>>(xcur, cums, gmean, gstd, Bseg, D);
    gn_apply_kernel<<<(int)(((long)N * D + 255) / 256), 256, 0, stream>>>(
        xcur, cums, gmean, gstd, gamma + (size_t)l * D, beta_g + (size_t)l * D, N, D, Bseg);

    // ping-pong xyz
    float* tmp = xyz_old; xyz_old = xyz_new; xyz_new = tmp;
  }

  hipMemcpyAsync(d_out, xcur, (size_t)N * D * 4, hipMemcpyDeviceToDevice, stream);
}
